// MatrixAttention_76149770158251
// MI455X (gfx1250) — matrix operand order repaired and checked by reading, not compiled, not run on hardware
//
#include <hip/hip_runtime.h>
#include <hip/hip_bf16.h>

typedef __attribute__((ext_vector_type(16))) __bf16 v16bf;
typedef __attribute__((ext_vector_type(4)))  __bf16 v4bf;
typedef __attribute__((ext_vector_type(8)))  float  v8f;

#define BATCH 8
#define N1D   4096
#define N2D   4096
#define DDIM  64
#define BM    128
#define BN    128
#define SROWB 72   // padded bf16 row stride: 144B -> rows start at bank 36r mod 64 (16 distinct), 16B aligned

__global__ __launch_bounds__(256)
void bmm_abT_bf16x3_wmma(const float* __restrict__ A,
                         const float* __restrict__ B,
                         float* __restrict__ C)
{
    // Dynamic LDS: 4 pre-split bf16 planes, (BM+BN)*SROWB*2 elems = 73,728 B.
    extern __shared__ __bf16 smem[];
    __bf16* AsH = smem;
    __bf16* AsL = AsH + BM * SROWB;
    __bf16* BsH = AsL + BM * SROWB;
    __bf16* BsL = BsH + BN * SROWB;

    const int tid  = threadIdx.x;
    const int lane = tid & 31;
    const int w    = tid >> 5;       // wave 0..7
    const int wr   = w >> 1;         // 0..3 -> 32-row strip
    const int wc   = w & 1;          // 0..1 -> 64-col strip
    const int bz   = blockIdx.z;
    const int row0 = blockIdx.y * BM;
    const int col0 = blockIdx.x * BN;

    // ---------------- global -> split-bf16 LDS planes (convert once per block) ----------------
    // A: K-chunks of 8 stored in natural order so each lane's 16-elem
    //    WMMA fragment is one contiguous 32B LDS read.
    {
        const float4* ga = reinterpret_cast<const float4*>(A + ((size_t)bz * N1D + row0) * DDIM);
        #pragma unroll
        for (int i = 0; i < (BM * DDIM / 4) / 256; ++i) {
            int fi  = tid + i * 256;           // float4 index within block
            int r   = fi >> 4;                 // 16 float4 per 64-elem row
            int col = (fi & 15) << 2;          // element column, multiple of 4
            int ch  = col >> 3;                // K-chunk 0..7
            int chp = ch;  // natural K order (no permutation)
            int colp = (chp << 3) | (col & 7);
            float4 q = ga[fi];
            float  f[4] = {q.x, q.y, q.z, q.w};
            v4bf hi, lo;
            #pragma unroll
            for (int e = 0; e < 4; ++e) {
                __bf16 hv = (__bf16)f[e];
                hi[e] = hv;
                lo[e] = (__bf16)(f[e] - (float)hv);
            }
            *reinterpret_cast<v4bf*>(&AsH[r * SROWB + colp]) = hi;
            *reinterpret_cast<v4bf*>(&AsL[r * SROWB + colp]) = lo;
        }
        // B: natural layout (fragment = 16 consecutive K per lane).
        const float4* gb = reinterpret_cast<const float4*>(B + ((size_t)bz * N2D + col0) * DDIM);
        #pragma unroll
        for (int i = 0; i < (BN * DDIM / 4) / 256; ++i) {
            int fi  = tid + i * 256;
            int r   = fi >> 4;
            int col = (fi & 15) << 2;
            float4 q = gb[fi];
            float  f[4] = {q.x, q.y, q.z, q.w};
            v4bf hi, lo;
            #pragma unroll
            for (int e = 0; e < 4; ++e) {
                __bf16 hv = (__bf16)f[e];
                hi[e] = hv;
                lo[e] = (__bf16)(f[e] - (float)hv);
            }
            *reinterpret_cast<v4bf*>(&BsH[r * SROWB + col]) = hi;
            *reinterpret_cast<v4bf*>(&BsL[r * SROWB + col]) = lo;
        }
    }
    __syncthreads();

    const int l = lane & 15;
    const int h = lane >> 4;
    const int fragOff = h * 16;      // this lane-half's contiguous fragment offset

    v8f acc[2][4] = {};

    #pragma unroll
    for (int ks = 0; ks < 2; ++ks) {
        const int kOff = ks * 32 + fragOff;

        // A fragments (natural order): lanes 0-15 K = k0..k0+15,
        // lanes 16-31 K = k0+16..31 — one contiguous 32B read per plane.
        v16bf aH[2], aL[2];
        #pragma unroll
        for (int mt = 0; mt < 2; ++mt) {
            const int arow = wr * 32 + mt * 16 + l;
            aH[mt] = *reinterpret_cast<const v16bf*>(&AsH[arow * SROWB + kOff]);
            aL[mt] = *reinterpret_cast<const v16bf*>(&AsL[arow * SROWB + kOff]);
        }

        // B fragments: lanes 0-15 K = k0..k0+15 (col = lane), lanes 16-31 K = k0+16..31.
        v16bf bH[4], bL[4];
        #pragma unroll
        for (int nt = 0; nt < 4; ++nt) {
            const int brow = wc * 64 + nt * 16 + l;
            bH[nt] = *reinterpret_cast<const v16bf*>(&BsH[brow * SROWB + kOff]);
            bL[nt] = *reinterpret_cast<const v16bf*>(&BsL[brow * SROWB + kOff]);
        }

        // 3-term bf16 split accumulate (drop lo*lo, ~2^-16 relative).
        #pragma unroll
        for (int mt = 0; mt < 2; ++mt) {
            #pragma unroll
            for (int nt = 0; nt < 4; ++nt) {
                acc[mt][nt] = __builtin_amdgcn_wmma_f32_16x16x32_bf16(
                    false, aH[mt], false, bH[nt], (short)0, acc[mt][nt], false, false);
                acc[mt][nt] = __builtin_amdgcn_wmma_f32_16x16x32_bf16(
                    false, aH[mt], false, bL[nt], (short)0, acc[mt][nt], false, false);
                acc[mt][nt] = __builtin_amdgcn_wmma_f32_16x16x32_bf16(
                    false, aL[mt], false, bH[nt], (short)0, acc[mt][nt], false, false);
            }
        }
    }

    // ---------------- store: VGPR v holds M = v + 8*(lane/16), N = lane%16 ----------------
    #pragma unroll
    for (int mt = 0; mt < 2; ++mt) {
        #pragma unroll
        for (int nt = 0; nt < 4; ++nt) {
            const int gr0 = row0 + wr * 32 + mt * 16 + h * 8;
            const int gc  = col0 + wc * 64 + nt * 16 + l;
            float* cp = C + ((size_t)bz * N1D + gr0) * (size_t)N2D + gc;
            #pragma unroll
            for (int vv = 0; vv < 8; ++vv)
                __builtin_nontemporal_store(acc[mt][nt][vv], cp + (size_t)vv * N2D);
        }
    }
}

extern "C" void kernel_launch(void* const* d_in, const int* in_sizes, int n_in,
                              void* d_out, int out_size, void* d_ws, size_t ws_size,
                              hipStream_t stream) {
    (void)in_sizes; (void)n_in; (void)out_size; (void)d_ws; (void)ws_size;
    const float* A = (const float*)d_in[0];   // [8, 4096, 64] f32
    const float* B = (const float*)d_in[1];   // [8, 4096, 64] f32
    float*       C = (float*)d_out;           // [8, 4096, 4096] f32

    const size_t shmem = (size_t)(BM + BN) * SROWB * 2 * sizeof(__bf16);  // 73,728 B
    dim3 block(256, 1, 1);
    dim3 grid(N2D / BN, N1D / BM, BATCH);     // (32, 32, 8)
    bmm_abT_bf16x3_wmma<<<grid, block, shmem, stream>>>(A, B, C);
}
